// DistanceEnergyModel_13597866459755
// MI455X (gfx1250) — compile-verified, matrix operand order checked by reading, not run on hardware
//
#include <hip/hip_runtime.h>
#include <hip/hip_bf16.h>

typedef __attribute__((ext_vector_type(16))) _Float16 v16h;
typedef __attribute__((ext_vector_type(8)))  float    v8f;

#define H 512
#define NT 32            // 512 / 16 N-tiles
#define KT2 16           // 512 / 32 K-tiles for H-dim GEMMs
#define MT 2             // M-tiles (32 edges) per block

// ---- dynamic LDS layout (in _Float16 elements) ----
#define SH_H0     0                        // MT x 512      (16x32 input A-tiles)
#define SH_A1     1024                     // MT x KT2 x 512 (a1 fwd, dz1 bwd)
#define SH_DZ2    (SH_A1 + MT * KT2 * 512) // MT x KT2 x 512
#define SH_DSIG1  (SH_DZ2 + MT * KT2 * 512)// MT x NT x 256  (silu'(z1), C layout)
#define SH_TOTAL  (SH_DSIG1 + MT * NT * 256)
#define LDS_BYTES (SH_TOTAL * 2)

// ---- workspace layout (in _Float16 elements) ----
#define WS_W1_SZ   (1 * NT * 512)      // 16384
#define WS_W2_SZ   (KT2 * NT * 512)    // 262144
#define WS_W2T_SZ  (KT2 * NT * 512)
#define WS_W1T_SZ  (KT2 * 1 * 512)     // 8192
#define OFF_PW1   0
#define OFF_PW2   (OFF_PW1  + WS_W1_SZ)
#define OFF_PW2T  (OFF_PW2  + WS_W2_SZ)
#define OFF_PW1T  (OFF_PW2T + WS_W2T_SZ)
#define OFF_TW1   (OFF_PW1T + WS_W1T_SZ)
#define OFF_TW2   (OFF_TW1  + WS_W1_SZ)
#define OFF_TW2T  (OFF_TW2  + WS_W2_SZ)
#define OFF_TW1T  (OFF_TW2T + WS_W2T_SZ)
#define WS_TOTAL_HALVES (OFF_TW1T + WS_W1T_SZ)

// fast sigmoid: single v_exp_f32 + v_rcp_f32 (no IEEE divide chain)
__device__ __forceinline__ float sigmf(float x) {
    return __builtin_amdgcn_rcpf(1.0f + __expf(-x));
}

// Pack a row-major f32 weight (Ksrc x Nsrc, leading dim ld) into f16 WMMA-tile
// layout: block b = kt*NTb + nt; element (lane, j) -> k = kt*32 + j + 16*(lane>>4),
// n = nt*16 + (lane&15).  transpose=1 -> src[n*ld+k].
__global__ void pack_weights_kernel(_Float16* __restrict__ dst, const float* __restrict__ src,
                                    int KTb, int NTb, int Ksrc, int Nsrc, int ld, int transpose) {
    int tid = blockIdx.x * blockDim.x + threadIdx.x;
    int total = KTb * NTb * 512;
    if (tid >= total) return;
    int b    = tid >> 9;
    int r    = tid & 511;
    int lane = r >> 4;
    int j    = r & 15;
    int kt = b / NTb, nt = b % NTb;
    int k = kt * 32 + j + 16 * (lane >> 4);
    int n = nt * 16 + (lane & 15);
    float v = 0.0f;
    if (k < Ksrc && n < Nsrc)
        v = transpose ? src[n * ld + k] : src[k * ld + n];
    dst[tid] = (_Float16)v;
}

__global__ void zero_kernel(float* __restrict__ p, int n) {
    int i = blockIdx.x * blockDim.x + threadIdx.x;
    if (i < n) p[i] = 0.0f;
}

__global__ void finalize_kernel(float* __restrict__ out, const float* __restrict__ pb3,
                                const float* __restrict__ tb3, int e2, int e3) {
    if (blockIdx.x == 0 && threadIdx.x == 0)
        out[0] += (float)e2 * pb3[0] + (float)e3 * tb3[0];
}

// One block = 32 edges (2 M-tiles), 256 threads = 8 wave32.
// Each wave owns 4 N-tiles of H=512 and both M-tiles (B-tile loads amortized 2x).
__global__ __launch_bounds__(256)
void energy_mlp_kernel(const float* __restrict__ x, const float* __restrict__ sigma,
                       const int* __restrict__ edges, int E, int arity,
                       const _Float16* __restrict__ W1p, const _Float16* __restrict__ W2p,
                       const _Float16* __restrict__ W2Tp, const _Float16* __restrict__ W1Tp,
                       const float* __restrict__ b1, const float* __restrict__ b2,
                       const float* __restrict__ W3,
                       float* __restrict__ energy_out, float* __restrict__ gradx) {
    extern __shared__ _Float16 sm[];
    __shared__ int   sh_eidx[32 * 3];
    __shared__ float sh_epart[8];

    const int tid   = threadIdx.x;
    const int wave  = tid >> 5;
    const int lane  = tid & 31;
    const int ebase = blockIdx.x * (16 * MT);
    const int ncol  = lane & 15;
    const int mrow  = 8 * (lane >> 4);

    // ---------- build padded input tiles h0 (2 x 16 x 32) ----------
    if (tid < 16 * MT) {
        const int m  = tid;
        const int mt = m >> 4, ml = m & 15;
        _Float16* base = sm + SH_H0 + mt * 512;
        for (int j = 0; j < 16; ++j) { base[ml * 16 + j] = (_Float16)0.0f; base[256 + ml * 16 + j] = (_Float16)0.0f; }
        int e = ebase + m; if (e >= E) e = E - 1;
        const float sg = sigma[0];
        for (int a = 0; a < arity; ++a) {
            int idx = edges[e * arity + a];
            sh_eidx[m * 3 + a] = idx;
            base[ml * 16 + 2 * a + 0] = (_Float16)x[idx * 2 + 0];
            base[ml * 16 + 2 * a + 1] = (_Float16)x[idx * 2 + 1];
            base[ml * 16 + 2 * arity + a] = (_Float16)sg;
        }
    }
    __syncthreads();

    // ---------- layer 1: z1 = h0 @ W1 + b1 ; a1 = silu(z1) ----------
    {
        v16h aIn0 = *(const v16h*)(sm + SH_H0 + 0 * 512 + lane * 16);
        v16h aIn1 = *(const v16h*)(sm + SH_H0 + 1 * 512 + lane * 16);
        for (int i = 0; i < 4; ++i) {
            const int nt = wave * 4 + i;
            v16h b = *(const v16h*)(W1p + (size_t)nt * 512 + lane * 16);
            v8f c[MT]; c[0] = {}; c[1] = {};
            c[0] = __builtin_amdgcn_wmma_f32_16x16x32_f16(false, aIn0, false, b, (short)0, c[0], false, false);
            c[1] = __builtin_amdgcn_wmma_f32_16x16x32_f16(false, aIn1, false, b, (short)0, c[1], false, false);
            const int   n    = nt * 16 + ncol;
            const float bias = b1[n];
            const int kk = n & 31, kt = n >> 5;
#pragma unroll
            for (int mt = 0; mt < MT; ++mt)
#pragma unroll
                for (int v = 0; v < 8; ++v) {
                    float z  = c[mt][v] + bias;
                    float s  = sigmf(z);
                    float a  = z * s;
                    float ds = s * (1.0f + z * (1.0f - s));
                    int m = v + mrow;
                    sm[SH_A1 + (mt * KT2 + kt) * 512 + (m + 16 * (kk >> 4)) * 16 + (kk & 15)] = (_Float16)a;
                    sm[SH_DSIG1 + (mt * NT + nt) * 256 + v * 32 + lane] = (_Float16)ds;
                }
        }
    }
    __syncthreads();

    // ---------- layer 2 fwd + energy + dz2 ----------
    float eacc = 0.0f;
    for (int i = 0; i < 4; ++i) {
        const int nt = wave * 4 + i;
        v8f c[MT]; c[0] = {}; c[1] = {};
        for (int kt = 0; kt < KT2; ++kt) {
            const _Float16* bp = W2p + ((size_t)kt * NT + nt) * 512;
            __builtin_prefetch(W2p + ((size_t)(kt + 1) * NT + nt) * 512, 0, 0);
            v16h b  = *(const v16h*)(bp + lane * 16);
            v16h a0 = *(const v16h*)(sm + SH_A1 + (0 * KT2 + kt) * 512 + lane * 16);
            v16h a1 = *(const v16h*)(sm + SH_A1 + (1 * KT2 + kt) * 512 + lane * 16);
            c[0] = __builtin_amdgcn_wmma_f32_16x16x32_f16(false, a0, false, b, (short)0, c[0], false, false);
            c[1] = __builtin_amdgcn_wmma_f32_16x16x32_f16(false, a1, false, b, (short)0, c[1], false, false);
        }
        const int   n    = nt * 16 + ncol;
        const float bias = b2[n];
        const float w3   = W3[n];
        const int kk = n & 31, kt = n >> 5;
#pragma unroll
        for (int mt = 0; mt < MT; ++mt)
#pragma unroll
            for (int v = 0; v < 8; ++v) {
                float z  = c[mt][v] + bias;
                float s  = sigmf(z);
                float a2 = z * s;
                float ds = s * (1.0f + z * (1.0f - s));
                int  m     = v + mrow;
                bool valid = (ebase + mt * 16 + m) < E;
                eacc += valid ? a2 * w3 : 0.0f;
                float dz2 = valid ? w3 * ds : 0.0f;
                sm[SH_DZ2 + (mt * KT2 + kt) * 512 + (m + 16 * (kk >> 4)) * 16 + (kk & 15)] = (_Float16)dz2;
            }
    }
    __syncthreads();   // dz2 complete; SH_A1 free for reuse

    // ---------- da1 = dz2 @ W2^T ; dz1 = da1 * silu'(z1) ----------
    for (int i = 0; i < 4; ++i) {
        const int nt = wave * 4 + i;
        v8f c[MT]; c[0] = {}; c[1] = {};
        for (int kt = 0; kt < KT2; ++kt) {
            v16h b  = *(const v16h*)(W2Tp + ((size_t)kt * NT + nt) * 512 + lane * 16);
            v16h a0 = *(const v16h*)(sm + SH_DZ2 + (0 * KT2 + kt) * 512 + lane * 16);
            v16h a1 = *(const v16h*)(sm + SH_DZ2 + (1 * KT2 + kt) * 512 + lane * 16);
            c[0] = __builtin_amdgcn_wmma_f32_16x16x32_f16(false, a0, false, b, (short)0, c[0], false, false);
            c[1] = __builtin_amdgcn_wmma_f32_16x16x32_f16(false, a1, false, b, (short)0, c[1], false, false);
        }
        const int n  = nt * 16 + ncol;
        const int kk = n & 31, kt = n >> 5;
#pragma unroll
        for (int mt = 0; mt < MT; ++mt)
#pragma unroll
            for (int v = 0; v < 8; ++v) {
                int   m   = v + mrow;
                float ds1 = (float)sm[SH_DSIG1 + (mt * NT + nt) * 256 + v * 32 + lane];
                float dz1 = c[mt][v] * ds1;
                sm[SH_A1 + (mt * KT2 + kt) * 512 + (m + 16 * (kk >> 4)) * 16 + (kk & 15)] = (_Float16)dz1;
            }
    }
    __syncthreads();

    // ---------- dh0 = dz1 @ W1^T ; scatter grad(x) (waves 0,1 = M-tiles 0,1) ----------
    if (wave < MT) {
        const int mt = wave;
        v8f c = {};
        for (int kt = 0; kt < KT2; ++kt) {
            v16h a = *(const v16h*)(sm + SH_A1 + (mt * KT2 + kt) * 512 + lane * 16);
            v16h b = *(const v16h*)(W1Tp + (size_t)kt * 512 + lane * 16);
            c = __builtin_amdgcn_wmma_f32_16x16x32_f16(false, a, false, b, (short)0, c, false, false);
        }
        const int col = ncol;
#pragma unroll
        for (int v = 0; v < 8; ++v) {
            int m = v + mrow;
            if (col < 2 * arity && (ebase + mt * 16 + m) < E) {
                int obj = col >> 1, d = col & 1;
                int xi  = sh_eidx[(mt * 16 + m) * 3 + obj];
                atomicAdd(&gradx[xi * 2 + d], c[v]);
            }
        }
    }

    // ---------- energy reduction ----------
    for (int off = 16; off > 0; off >>= 1)
        eacc += __shfl_xor(eacc, off, 32);
    if (lane == 0) sh_epart[wave] = eacc;
    __syncthreads();
    if (tid == 0) {
        float t = 0.0f;
        for (int w = 0; w < 8; ++w) t += sh_epart[w];
        atomicAdd(energy_out, t);
    }
}

extern "C" void kernel_launch(void* const* d_in, const int* in_sizes, int n_in,
                              void* d_out, int out_size, void* d_ws, size_t ws_size,
                              hipStream_t stream) {
    const float* x      = (const float*)d_in[0];
    const float* sigma  = (const float*)d_in[1];
    const int*   e2ptr  = (const int*)d_in[2];
    const int*   e3ptr  = (const int*)d_in[3];
    const float* pW1 = (const float*)d_in[4];  const float* pb1 = (const float*)d_in[5];
    const float* pW2 = (const float*)d_in[6];  const float* pb2 = (const float*)d_in[7];
    const float* pW3 = (const float*)d_in[8];  const float* pb3 = (const float*)d_in[9];
    const float* tW1 = (const float*)d_in[10]; const float* tb1 = (const float*)d_in[11];
    const float* tW2 = (const float*)d_in[12]; const float* tb2 = (const float*)d_in[13];
    const float* tW3 = (const float*)d_in[14]; const float* tb3 = (const float*)d_in[15];

    const int E2 = in_sizes[2] / 2;
    const int E3 = in_sizes[3] / 3;

    if (ws_size < (size_t)WS_TOTAL_HALVES * sizeof(_Float16)) return;
    _Float16* ws = (_Float16*)d_ws;

    float* energy = (float*)d_out;
    float* gradx  = (float*)d_out + 1;

    zero_kernel<<<(out_size + 255) / 256, 256, 0, stream>>>((float*)d_out, out_size);

    auto packs = [&](const _Float16* dst, const float* src, int KTb, int NTb,
                     int Ksrc, int Nsrc, int ld, int tr) {
        int total = KTb * NTb * 512;
        pack_weights_kernel<<<(total + 255) / 256, 256, 0, stream>>>(
            (_Float16*)dst, src, KTb, NTb, Ksrc, Nsrc, ld, tr);
    };
    // pair (IN = 6)
    packs(ws + OFF_PW1,  pW1, 1,   NT, 6,   H, H, 0);
    packs(ws + OFF_PW2,  pW2, KT2, NT, H,   H, H, 0);
    packs(ws + OFF_PW2T, pW2, KT2, NT, H,   H, H, 1);
    packs(ws + OFF_PW1T, pW1, KT2, 1,  H,   6, H, 1);
    // triple (IN = 9)
    packs(ws + OFF_TW1,  tW1, 1,   NT, 9,   H, H, 0);
    packs(ws + OFF_TW2,  tW2, KT2, NT, H,   H, H, 0);
    packs(ws + OFF_TW2T, tW2, KT2, NT, H,   H, H, 1);
    packs(ws + OFF_TW1T, tW1, KT2, 1,  H,   9, H, 1);

    const int EB = 16 * MT;   // 32 edges per block

    energy_mlp_kernel<<<(E2 + EB - 1) / EB, 256, LDS_BYTES, stream>>>(
        x, sigma, e2ptr, E2, 2,
        ws + OFF_PW1, ws + OFF_PW2, ws + OFF_PW2T, ws + OFF_PW1T,
        pb1, pb2, pW3, energy, gradx);

    energy_mlp_kernel<<<(E3 + EB - 1) / EB, 256, LDS_BYTES, stream>>>(
        x, sigma, e3ptr, E3, 3,
        ws + OFF_TW1, ws + OFF_TW2, ws + OFF_TW2T, ws + OFF_TW1T,
        tb1, tb2, tW3, energy, gradx);

    finalize_kernel<<<1, 32, 0, stream>>>(energy, pb3, tb3, E2, E3);
}
